// SelfAttentionHead_38156489458325
// MI455X (gfx1250) — matrix operand order repaired and checked by reading, compiled, not run on hardware
//
#include <hip/hip_runtime.h>

// ---------------------------------------------------------------------------
// Causal self-attention head, bf16 WMMA pipeline for gfx1250 (wave32).
//   B=8, T=2048, DM=1024, DK=DV=128
// Stages: f32->bf16 convert (scale folded into Wq), QKV projection (WMMA,
// V stored transposed), flash attention (WMMA; K tiles staged double-buffered
// into LDS by the Tensor Data Mover with hardware row padding).
// ---------------------------------------------------------------------------

typedef __attribute__((ext_vector_type(16))) __bf16 v16bf;
typedef __attribute__((ext_vector_type(8)))  __bf16 v8bf;
typedef __attribute__((ext_vector_type(8)))  float  v8f;
typedef __attribute__((ext_vector_type(4)))  float  v4f;
typedef __attribute__((ext_vector_type(4)))  unsigned int u32x4;
typedef __attribute__((ext_vector_type(8)))  int  i32x8;
typedef __attribute__((ext_vector_type(4)))  int  i32x4;

constexpr int B  = 8;
constexpr int T  = 2048;
constexpr int DM = 1024;
constexpr int DK = 128;
constexpr int DV = 128;

#if defined(__AMDGCN__) && __has_builtin(__builtin_amdgcn_tensor_load_to_lds) && \
    __has_builtin(__builtin_amdgcn_s_wait_tensorcnt)
#define ATTN_USE_TDM 1
#else
#define ATTN_USE_TDM 0
#endif

static __device__ __forceinline__ v16bf cat8(v8bf a, v8bf b) {
  return __builtin_shufflevector(a, b, 0,1,2,3,4,5,6,7,8,9,10,11,12,13,14,15);
}

#if ATTN_USE_TDM
typedef __attribute__((address_space(3))) const char* lds_cptr_t;
static __device__ __forceinline__ unsigned int lds_addr_of(const void* p) {
  return (unsigned int)(size_t)(lds_cptr_t)p;
}

// Issue a TDM load of a 32x128 bf16 tile (rows t, contiguous dim d) into LDS,
// with hardware padding of 4 DWORDs (16B) after every 64 DWORDs (one 256B row)
// so the LDS image has row stride 272B = 136 bf16 (bank-conflict-free reads).
static __device__ __forceinline__ void tdm_load_k_tile(const __bf16* gsrc,
                                                       unsigned int lds_off) {
  unsigned long long ga = (unsigned long long)(size_t)gsrc;
  u32x4 g0;
  g0[0] = 1u;                                   // count=1 valid descriptor
  g0[1] = lds_off;                              // lds_addr (bytes)
  g0[2] = (unsigned int)ga;                     // global_addr[31:0]
  g0[3] = (unsigned int)(ga >> 32) | (2u << 30);// global_addr[56:32] | type=2
  i32x8 g1;
  g1[0] = (1 << 16)      // data_size = 2 bytes
        | (1 << 20)      // pad_enable
        | (5 << 22)      // pad_interval: 64 DWORDs (one 256B row)
        | (3 << 25);     // pad_amount: 4 DWORDs (16B)
  g1[1] = (int)(128u << 16);   // tensor_dim0[15:0] = 128 (elements per row)
  g1[2] = (int)(32u  << 16);   // tensor_dim1[15:0] = 32 rows
  g1[3] = (int)(128u << 16);   // tile_dim0 = 128
  g1[4] = 32;                  // tile_dim1 = 32
  g1[5] = 128;                 // tensor_dim0_stride = 128 elements
  g1[6] = 0;
  g1[7] = 0;
  i32x4 gz = {0, 0, 0, 0};     // 2D tile: groups 2/3 unused
#if __clang_major__ >= 23
  i32x8 gz8 = {0, 0, 0, 0, 0, 0, 0, 0};
  __builtin_amdgcn_tensor_load_to_lds(g0, g1, gz, gz, gz8, 0);
#else
  __builtin_amdgcn_tensor_load_to_lds(g0, g1, gz, gz, 0);
#endif
}
#endif  // ATTN_USE_TDM

// ---------------------------------------------------------------------------
// Stage 1: f32 -> bf16 conversion with optional premultiply (folds the
// attention 1/sqrt(DK) scale into Wq). 8 elements per thread.
// ---------------------------------------------------------------------------
__global__ void cvt_f32_bf16_kernel(const float* __restrict__ in,
                                    __bf16* __restrict__ out, int n8,
                                    float premul) {
  int i = blockIdx.x * blockDim.x + threadIdx.x;
  if (i < n8) {
    v4f f0 = ((const v4f*)in)[2 * i + 0];
    v4f f1 = ((const v4f*)in)[2 * i + 1];
    v8bf o;
#pragma unroll
    for (int j = 0; j < 4; ++j) {
      o[j]     = (__bf16)(f0[j] * premul);
      o[4 + j] = (__bf16)(f1[j] * premul);
    }
    ((v8bf*)out)[i] = o;
  }
}

// ---------------------------------------------------------------------------
// Stage 2: QKV projection.  q[t,d] = sum_c x[t,c] * W[d,c]  (x @ W^T)
// grid.x = B*T/16 row tiles, grid.y = 3 (q,k,v); 256 threads = 8 waves,
// wave w computes d-tile w. Q,K stored row-major [b][t][d]; V stored
// TRANSPOSED as VT[b][d][t] so the attention PV B-operand is a contiguous
// 32B load (and the C-fragment rows become one packed v8bf store per lane).
// ---------------------------------------------------------------------------
__global__ __launch_bounds__(256)
void qkv_proj_kernel(const __bf16* __restrict__ xb,
                     const __bf16* __restrict__ wq,
                     const __bf16* __restrict__ wk,
                     const __bf16* __restrict__ wv,
                     __bf16* __restrict__ qb,
                     __bf16* __restrict__ kb,
                     __bf16* __restrict__ vt) {
  const int wave = threadIdx.x >> 5;
  const int lane = threadIdx.x & 31;
  const int half = lane >> 4;
  const int l16  = lane & 15;
  const int row0 = blockIdx.x * 16;                // tile of 16 rows in (B*T)
  const __bf16* W = (blockIdx.y == 0) ? wq : (blockIdx.y == 1) ? wk : wv;
  const int d0 = wave * 16;

  const __bf16* arow = xb + (size_t)(row0 + l16) * DM + half * 16;
  const __bf16* brow = W  + (size_t)(d0  + l16) * DM + half * 16;

  v8f acc = {};
#pragma unroll 4
  for (int c = 0; c < DM; c += 32) {
    v16bf a = cat8(*(const v8bf*)(arow + c), *(const v8bf*)(arow + c + 8));
    v16bf b = *(const v16bf*)(brow + c);
    acc = __builtin_amdgcn_wmma_f32_16x16x32_bf16(false, a, false, b,
                                                  (short)0, acc, false, false);
  }

  if (blockIdx.y == 2) {
    const int bidx = row0 / T;
    const int t0   = row0 - bidx * T;
    v8bf pk;
#pragma unroll
    for (int r = 0; r < 8; ++r) pk[r] = (__bf16)acc[r];
    *(v8bf*)(vt + ((size_t)bidx * DV + d0 + l16) * T + t0 + half * 8) = pk;
  } else {
    __bf16* out = (blockIdx.y == 0) ? qb : kb;
#pragma unroll
    for (int r = 0; r < 8; ++r)
      out[(size_t)(row0 + r + half * 8) * DK + d0 + l16] = (__bf16)acc[r];
  }
}

// ---------------------------------------------------------------------------
// Stage 3: flash attention. One wave per (query-tile of 16, batch).
// Keys in chunks of 32; K tiles DMA'd into LDS by the TDM, double-buffered
// behind compute, synchronized with s_wait_tensorcnt. Online softmax in f32,
// P staged through a 1.3KB LDS tile (C->A cross-lane transpose), V read from
// transposed VT (contiguous B-fragments, L2-resident).
// ---------------------------------------------------------------------------
__global__ __launch_bounds__(32)
void flash_attn_kernel(const __bf16* __restrict__ qb,
                       const __bf16* __restrict__ kb,
                       const __bf16* __restrict__ vt,
                       float* __restrict__ out) {
  __shared__ __align__(32) __bf16 Ps[16][40];       // 16 rows x 32 keys, padded
#if ATTN_USE_TDM
  __shared__ __align__(32) __bf16 Ks[2][32][136];   // 2 x (32 keys x 128+pad)
#endif

  const int lane = threadIdx.x;
  const int half = lane >> 4;
  const int l16  = lane & 15;
  const int tq   = blockIdx.x * 16;                 // first query row of tile
  const int b    = blockIdx.y;

  const __bf16* Q  = qb + ((size_t)b * T + tq) * DK;
  const __bf16* K  = kb + (size_t)b * T * DK;
  const __bf16* VT = vt + (size_t)b * DV * T;

  // Q tile as 4 A-fragments (K = 128 channels = 4 chunks of 32), in regs.
  v16bf qa[4];
#pragma unroll
  for (int c = 0; c < 4; ++c) {
    const __bf16* p = Q + (size_t)l16 * DK + c * 32 + half * 16;
    qa[c] = cat8(*(const v8bf*)p, *(const v8bf*)(p + 8));
  }

  float mrow[8], lrow[8];
  v8f o[8];
#pragma unroll
  for (int r = 0; r < 8; ++r) { mrow[r] = -__builtin_inff(); lrow[r] = 0.0f; }
#pragma unroll
  for (int vc = 0; vc < 8; ++vc) o[vc] = (v8f){};

  const int nchunk = (tq + 16 + 31) / 32;           // keys [0, tq+16) covered

#if ATTN_USE_TDM
  tdm_load_k_tile(K, lds_addr_of(&Ks[0][0][0]));    // prologue: chunk 0
#endif

  for (int ch = 0; ch < nchunk; ++ch) {
    const int k0 = ch * 32;

#if ATTN_USE_TDM
    if (ch + 1 < nchunk) {                          // double-buffer next chunk
      tdm_load_k_tile(K + (size_t)(k0 + 32) * DK,
                      lds_addr_of(&Ks[(ch + 1) & 1][0][0]));
      __builtin_amdgcn_s_wait_tensorcnt(1);         // current buffer ready
    } else {
      __builtin_amdgcn_s_wait_tensorcnt(0);
    }
    const __bf16* Kc = &Ks[ch & 1][0][0];
#endif
    if (k0 + 64 <= T)                               // VT line hint for L2
      __builtin_prefetch(VT + (size_t)l16 * T + k0 + 32, 0, 3);

    // S = Q * K^T over 32 keys: two accumulators (key groups of 16).
    v8f s0 = {}, s1 = {};
#pragma unroll
    for (int c = 0; c < 4; ++c) {
#if ATTN_USE_TDM
      const __bf16* p0 = Kc + (size_t)(l16)      * 136 + c * 32 + half * 16;
      const __bf16* p1 = Kc + (size_t)(l16 + 16) * 136 + c * 32 + half * 16;
      v16bf b0 = cat8(*(const v8bf*)p0, *(const v8bf*)(p0 + 8));
      v16bf b1 = cat8(*(const v8bf*)p1, *(const v8bf*)(p1 + 8));
#else
      v16bf b0 = *(const v16bf*)(K + (size_t)(k0 + l16) * DK + c * 32 + half * 16);
      v16bf b1 = *(const v16bf*)(K + (size_t)(k0 + 16 + l16) * DK + c * 32 + half * 16);
#endif
      s0 = __builtin_amdgcn_wmma_f32_16x16x32_bf16(false, qa[c], false, b0,
                                                   (short)0, s0, false, false);
      s1 = __builtin_amdgcn_wmma_f32_16x16x32_bf16(false, qa[c], false, b1,
                                                   (short)0, s1, false, false);
    }

    __syncthreads();  // previous iteration's Ps reads done before overwrite

    // Causal mask + online softmax. (1/sqrt(DK) already folded into Wq.)
    float scl[8];
#pragma unroll
    for (int r = 0; r < 8; ++r) {
      const int M  = tq + r + half * 8;             // global query row
      const int n0 = k0 + l16;                      // key of s0 column
      const int n1 = k0 + 16 + l16;                 // key of s1 column
      float a0 = (n0 <= M) ? s0[r] : -__builtin_inff();
      float a1 = (n1 <= M) ? s1[r] : -__builtin_inff();
      float mx = fmaxf(a0, a1);
#pragma unroll
      for (int off = 1; off < 16; off <<= 1) mx = fmaxf(mx, __shfl_xor(mx, off, 32));
      const float mnew = fmaxf(mrow[r], mx);        // mx finite: key k0 <= M always
      scl[r] = __expf(mrow[r] - mnew);
      float p0 = __expf(a0 - mnew);
      float p1 = __expf(a1 - mnew);
      float sum = p0 + p1;
#pragma unroll
      for (int off = 1; off < 16; off <<= 1) sum += __shfl_xor(sum, off, 32);
      lrow[r] = lrow[r] * scl[r] + sum;
      mrow[r] = mnew;
      Ps[r + half * 8][l16]      = (__bf16)p0;
      Ps[r + half * 8][l16 + 16] = (__bf16)p1;
    }

    // Rescale running O by exp(m_old - m_new).
#pragma unroll
    for (int vc = 0; vc < 8; ++vc)
#pragma unroll
      for (int r = 0; r < 8; ++r) o[vc][r] *= scl[r];

    __syncthreads();  // Ps writes visible before re-read in A layout

    // P as A-fragment (M=query row l16, K=32 keys).
    v16bf pa;
    {
      const __bf16* pr = &Ps[l16][half * 16];
      pa = cat8(*(const v8bf*)pr, *(const v8bf*)(pr + 8));
    }

    // O += P * V : B-frag = 16 consecutive keys at fixed vdim, contiguous in VT.
#pragma unroll
    for (int vc = 0; vc < 8; ++vc) {
      v16bf vfrag = *(const v16bf*)(VT + (size_t)(vc * 16 + l16) * T + k0 + half * 16);
      o[vc] = __builtin_amdgcn_wmma_f32_16x16x32_bf16(false, pa, false, vfrag,
                                                      (short)0, o[vc], false, false);
    }
  }

  // Normalize and write fp32 output.
#pragma unroll
  for (int vc = 0; vc < 8; ++vc)
#pragma unroll
    for (int r = 0; r < 8; ++r) {
      const int M = r + half * 8;
      out[((size_t)b * T + tq + M) * DV + vc * 16 + l16] = o[vc][r] / lrow[r];
    }
}

// ---------------------------------------------------------------------------
// Host-side launcher
// ---------------------------------------------------------------------------
extern "C" void kernel_launch(void* const* d_in, const int* in_sizes, int n_in,
                              void* d_out, int out_size, void* d_ws, size_t ws_size,
                              hipStream_t stream) {
  (void)in_sizes; (void)n_in; (void)out_size; (void)ws_size;

  const float* x  = (const float*)d_in[0];
  const float* Wq = (const float*)d_in[1];
  const float* Wk = (const float*)d_in[2];
  const float* Wv = (const float*)d_in[3];
  float* out = (float*)d_out;

  // Workspace carve-up (bf16), total ~44.7 MB.
  __bf16* p   = (__bf16*)d_ws;
  __bf16* xb  = p; p += (size_t)B * T * DM;
  __bf16* wqb = p; p += (size_t)DK * DM;
  __bf16* wkb = p; p += (size_t)DK * DM;
  __bf16* wvb = p; p += (size_t)DK * DM;
  __bf16* qb  = p; p += (size_t)B * T * DK;
  __bf16* kb  = p; p += (size_t)B * T * DK;
  __bf16* vt  = p; p += (size_t)B * DV * T;   // transposed V: [b][d][t]

  const float scale = 0.08838834764831843f;   // 1/sqrt(DK), folded into Wq

  {
    int n8 = (B * T * DM) / 8;
    cvt_f32_bf16_kernel<<<(n8 + 255) / 256, 256, 0, stream>>>(x, xb, n8, 1.0f);
    n8 = (DK * DM) / 8;
    cvt_f32_bf16_kernel<<<(n8 + 255) / 256, 256, 0, stream>>>(Wq, wqb, n8, scale);
    cvt_f32_bf16_kernel<<<(n8 + 255) / 256, 256, 0, stream>>>(Wk, wkb, n8, 1.0f);
    cvt_f32_bf16_kernel<<<(n8 + 255) / 256, 256, 0, stream>>>(Wv, wvb, n8, 1.0f);
  }

  qkv_proj_kernel<<<dim3(B * T / 16, 3), 256, 0, stream>>>(xb, wqb, wkb, wvb,
                                                           qb, kb, vt);

  flash_attn_kernel<<<dim3(T / 16, B), 32, 0, stream>>>(qb, kb, vt, out);
}
